// PlugAndPlayDCN_6201932775434
// MI455X (gfx1250) — compile-verified, matrix operand order checked by reading, not run on hardware
//
#include <hip/hip_runtime.h>
#include <hip/hip_bf16.h>
#include <math.h>

typedef __bf16 bf16;
typedef __attribute__((ext_vector_type(8)))  __bf16 v8bf;
typedef __attribute__((ext_vector_type(16))) __bf16 v16bf;
typedef __attribute__((ext_vector_type(8)))  float  v8f;
typedef __attribute__((ext_vector_type(4)))  unsigned int u32x4;
typedef __attribute__((ext_vector_type(4)))  int i32x4;
typedef __attribute__((ext_vector_type(8)))  int i32x8;

#define B_   4
#define C_   256
#define HW_  4096
#define NPT_ 147456   // B * K2 * HW  = 4*9*4096

#if defined(__AMDGCN__) && __has_builtin(__builtin_amdgcn_tensor_load_to_lds)
#define USE_TDM 1
#else
#define USE_TDM 0
#endif

// ---------------------------------------------------------------------------
// Kernel 1: NCHW -> NHWC transpose of x (fp32), LDS tiled, conflict-free pad.
// ---------------------------------------------------------------------------
__global__ void xpose_kernel(const float* __restrict__ x, float* __restrict__ xt) {
  __shared__ float tile[32][65];
  int bx = blockIdx.x;
  int ct = bx & 7, pt = (bx >> 3) & 63, b = bx >> 9;
  int c0 = ct * 32, p0 = pt * 64;
  const float* xb = x + (size_t)b * C_ * HW_;
  for (int i = threadIdx.x; i < 2048; i += 256) {
    int r = i >> 6, col = i & 63;                       // coalesced read along pixels
    tile[r][col] = xb[(size_t)(c0 + r) * HW_ + p0 + col];
  }
  __syncthreads();
  float* xtb = xt + (size_t)b * HW_ * C_;
  for (int i = threadIdx.x; i < 2048; i += 256) {
    int p = i >> 5, c = i & 31;                         // coalesced write along channels
    xtb[(size_t)(p0 + p) * C_ + c0 + c] = tile[c][p];
  }
}

// ---------------------------------------------------------------------------
// Kernel 2: weight (OC,C,3,3) fp32 -> bf16 A-matrix layout [o][k2*256 + c].
// ---------------------------------------------------------------------------
__global__ void wprep_kernel(const float* __restrict__ w, bf16* __restrict__ abf) {
  int gid = blockIdx.x * 256 + threadIdx.x;             // 256*2304 total
  int o = gid / 2304, r = gid - o * 2304;
  int k2 = r >> 8, c = r & 255;
  abf[gid] = (bf16)w[(o * 256 + c) * 9 + k2];
}

// ---------------------------------------------------------------------------
// Kernel 3: head conv (256->27) + offsets/mask -> per (b,k2,pix) 4 indices and
// 4 fused weights (bilinear * valid * mask). p_w reads are wave-uniform.
// ---------------------------------------------------------------------------
__global__ void head_kernel(const float* __restrict__ x, const float* __restrict__ em,
                            const float* __restrict__ pw, const float* __restrict__ pb,
                            int4* __restrict__ idxw, float4* __restrict__ wgtw) {
  int gid = blockIdx.x * 256 + threadIdx.x;             // one thread per (b,pix)
  int b = gid >> 12, pix = gid & 4095;
  int yc = pix >> 6, xc = pix & 63;
  float acc[27];
#pragma unroll
  for (int i = 0; i < 27; ++i) acc[i] = 0.f;
  const float* xb = x + (size_t)b * C_ * HW_;
  for (int c = 0; c < C_; ++c) {
    float xv[9];
#pragma unroll
    for (int t = 0; t < 9; ++t) {
      int yy = yc + (t / 3) - 1, xx = xc + (t % 3) - 1;
      xv[t] = (yy >= 0 && yy < 64 && xx >= 0 && xx < 64)
                  ? xb[(size_t)c * HW_ + (yy << 6) + xx] : 0.f;
    }
    const float* pwc = pw + c * 9;                      // uniform address -> s_load
#pragma unroll
    for (int oc = 0; oc < 27; ++oc) {
      const float* po = pwc + oc * 2304;
#pragma unroll
      for (int t = 0; t < 9; ++t) acc[oc] = fmaf(xv[t], po[t], acc[oc]);
    }
  }
  float emv = em[(size_t)b * HW_ + pix];
#pragma unroll
  for (int k2 = 0; k2 < 9; ++k2) {
    float dy = tanhf(acc[2 * k2]     + pb[2 * k2])     * 3.f;
    float dx = tanhf(acc[2 * k2 + 1] + pb[2 * k2 + 1]) * 3.f;
    float mm = 1.f / (1.f + expf(-(acc[18 + k2] + pb[18 + k2]))) + 0.5f * emv;
    float py = (float)(yc - 1 + k2 / 3) + dy;
    float px = (float)(xc - 1 + k2 % 3) + dx;
    float y0f = floorf(py), x0f = floorf(px);
    float wy = py - y0f, wx = px - x0f;
    int y0 = (int)y0f, x0 = (int)x0f;
    int4 id; float4 wt;
    {
      int yi = min(max(y0, 0), 63), xi = min(max(x0, 0), 63);
      float v = (y0 >= 0 && y0 <= 63 && x0 >= 0 && x0 <= 63) ? 1.f : 0.f;
      id.x = yi * 64 + xi; wt.x = (1.f - wy) * (1.f - wx) * mm * v;
    }
    {
      int yi = min(max(y0, 0), 63), xi = min(max(x0 + 1, 0), 63);
      float v = (y0 >= 0 && y0 <= 63 && x0 + 1 >= 0 && x0 + 1 <= 63) ? 1.f : 0.f;
      id.y = yi * 64 + xi; wt.y = (1.f - wy) * wx * mm * v;
    }
    {
      int yi = min(max(y0 + 1, 0), 63), xi = min(max(x0, 0), 63);
      float v = (y0 + 1 >= 0 && y0 + 1 <= 63 && x0 >= 0 && x0 <= 63) ? 1.f : 0.f;
      id.z = yi * 64 + xi; wt.z = wy * (1.f - wx) * mm * v;
    }
    {
      int yi = min(max(y0 + 1, 0), 63), xi = min(max(x0 + 1, 0), 63);
      float v = (y0 + 1 >= 0 && y0 + 1 <= 63 && x0 + 1 >= 0 && x0 + 1 <= 63) ? 1.f : 0.f;
      id.w = yi * 64 + xi; wt.w = wy * wx * mm * v;
    }
    int o = (b * 9 + k2) * HW_ + pix;
    idxw[o] = id; wgtw[o] = wt;
  }
}

// ---------------------------------------------------------------------------
// Kernel 4: build v[b][k2][pix][c] bf16. Wave per pixel-tap, lane = 8 channels.
// ---------------------------------------------------------------------------
__global__ void vbuild_kernel(const float* __restrict__ xt, const int4* __restrict__ idxw,
                              const float4* __restrict__ wgtw, bf16* __restrict__ v) {
  int tid = threadIdx.x;
  int wv = blockIdx.x * 8 + (tid >> 5);                 // (b*9+k2)*4096 + pix
  int lane = tid & 31;
  int b = wv / 36864;
  int4  id = idxw[wv];                                  // broadcast load
  float4 wt = wgtw[wv];
  const float* xb = xt + (size_t)b * HW_ * C_ + lane * 8;
  float acc[8] = {0.f, 0.f, 0.f, 0.f, 0.f, 0.f, 0.f, 0.f};
  const int   ix[4] = {id.x, id.y, id.z, id.w};
  const float ww[4] = {wt.x, wt.y, wt.z, wt.w};
#pragma unroll
  for (int j = 0; j < 4; ++j) {
    const float4* row = (const float4*)(xb + (size_t)ix[j] * C_);
    float4 r0 = row[0], r1 = row[1];
    acc[0] = fmaf(ww[j], r0.x, acc[0]); acc[1] = fmaf(ww[j], r0.y, acc[1]);
    acc[2] = fmaf(ww[j], r0.z, acc[2]); acc[3] = fmaf(ww[j], r0.w, acc[3]);
    acc[4] = fmaf(ww[j], r1.x, acc[4]); acc[5] = fmaf(ww[j], r1.y, acc[5]);
    acc[6] = fmaf(ww[j], r1.z, acc[6]); acc[7] = fmaf(ww[j], r1.w, acc[7]);
  }
  struct alignas(16) P8 { bf16 h[8]; } p;
#pragma unroll
  for (int i = 0; i < 8; ++i) p.h[i] = (bf16)acc[i];
  *(P8*)(v + (size_t)wv * 256 + lane * 8) = p;          // coalesced b128 store
}

// ---------------------------------------------------------------------------
// TDM helper: 2D tile load Global->LDS, data_size=2B, tile_d0 elems per row,
// row pitch stride_elems, LDS rows padded 64B data + 16B (pad_interval=16 dw,
// pad_amount=4 dw). Descriptor per CDNA5 ISA ch.8 (D# groups 0/1; 2-3 zero).
// ---------------------------------------------------------------------------
__device__ __forceinline__ void tdm_load_2d(unsigned lds_addr, const void* gaddr,
                                            unsigned tile_d0, unsigned tile_d1,
                                            unsigned stride_elems) {
#if USE_TDM
  unsigned long long ga = (unsigned long long)(size_t)gaddr;
  u32x4 g0 = {
      1u,                                               // count=1, user mode
      lds_addr,                                         // D#.lds_addr
      (unsigned)ga,                                     // global_addr[31:0]
      (((unsigned)(ga >> 32)) & 0x01FFFFFFu) | (2u << 30)  // addr[56:32] | type=2
  };
  const unsigned td = 0x40000000u;                      // huge tensor dims: no OOB clip
  i32x8 g1 = {
      (int)((1u << 16) | (1u << 20) | (3u << 22) | (3u << 25)),
      //     data_size=2B  pad_en      pad_int=16dw pad_amt=4dw ; wg_mask=0
      (int)((td & 0xFFFFu) << 16),                      // abar=0 | tensor_dim0 lo
      (int)((td >> 16) | ((td & 0xFFFFu) << 16)),       // tensor_dim0 hi | dim1 lo
      (int)((td >> 16) | (tile_d0 << 16)),              // tensor_dim1 hi | tile_dim0
      (int)(tile_d1),                                   // tile_dim1 | tile_dim2=0
      (int)(stride_elems),                              // tensor_dim0_stride[31:0]
      0,                                                // stride hi | dim1_stride lo
      0
  };
  i32x4 z4 = {0, 0, 0, 0};
#if __clang_major__ >= 23
  i32x8 z8 = {0, 0, 0, 0, 0, 0, 0, 0};
  __builtin_amdgcn_tensor_load_to_lds(g0, g1, z4, z4, z8, 0);
#else
  __builtin_amdgcn_tensor_load_to_lds(g0, g1, z4, z4, 0);
#endif
#else
  (void)lds_addr; (void)gaddr; (void)tile_d0; (void)tile_d1; (void)stride_elems;
#endif
}

// ---------------------------------------------------------------------------
// Kernel 5: WMMA GEMM. out[b,oc,pix] = sum_{k2,c} A[oc][k2*256+c] * v[b,k2,pix,c]
// Workgroup: 128 OC x 64 pixels; 8 waves, 2x2 16x16 subtiles per wave.
// K = 2304 in 72 chunks of 32. TDM path: wave 0 issues double-buffered
// tensor_load_to_lds DMA (TENSORcnt), other waves ride the barriers.
// ---------------------------------------------------------------------------
__global__ void __launch_bounds__(256, 2)
dcn_gemm_kernel(const bf16* __restrict__ abf, const bf16* __restrict__ v,
                const float* __restrict__ bias, float* __restrict__ out) {
#if USE_TDM
  __shared__ alignas(16) bf16 As[2][128 * 40];          // rows: 32 data + 8 pad bf16
  __shared__ alignas(16) bf16 Bs[2][64 * 40];
#else
  __shared__ alignas(16) bf16 As[1][128 * 40];
  __shared__ alignas(16) bf16 Bs[1][64 * 40];
#endif
  int tid = threadIdx.x;
  int bx = blockIdx.x;
  int mhalf = bx & 1, ptile = (bx >> 1) & 63, b = bx >> 7;
  int oc0 = mhalf * 128, pix0 = ptile * 64;
  int lane = tid & 31, wid = tid >> 5;
  int lm = lane & 15, lh = lane >> 4;
  int m0 = (wid & 3) * 2, n0 = (wid >> 2) * 2;          // subtile bases for this wave
  v8f acc[2][2] = {};
  const bf16* vb = v + (size_t)b * 9 * HW_ * 256;
  const bf16* ab = abf + (size_t)oc0 * 2304;

#if USE_TDM
  if (wid == 0) {                                       // prologue: chunk 0 -> buf 0
    tdm_load_2d((unsigned)(size_t)&As[0][0], ab, 32, 128, 2304);
    tdm_load_2d((unsigned)(size_t)&Bs[0][0], vb + (size_t)pix0 * 256, 32, 64, 256);
  }
  for (int kc = 0; kc < 72; ++kc) {
    int cur = kc & 1;
    __syncthreads();                                    // buf cur^1 free to overwrite
    if (wid == 0) {
      if (kc + 1 < 72) {
        int kn = kc + 1, k2n = kn >> 3, c0n = (kn & 7) * 32;
        tdm_load_2d((unsigned)(size_t)&As[cur ^ 1][0],
                    ab + (size_t)k2n * 256 + c0n, 32, 128, 2304);
        tdm_load_2d((unsigned)(size_t)&Bs[cur ^ 1][0],
                    vb + ((size_t)k2n * HW_ + pix0) * 256 + c0n, 32, 64, 256);
        __builtin_amdgcn_s_wait_tensorcnt(2);           // chunk kc's 2 DMAs landed
      } else {
        __builtin_amdgcn_s_wait_tensorcnt(0);
      }
    }
    __syncthreads();                                    // buf cur visible to all
#else
  for (int kc = 0; kc < 72; ++kc) {
    int cur = 0;
    int k2 = kc >> 3;
    int c0 = (kc & 7) * 32;
    const bf16* ag = ab + (size_t)k2 * 256 + c0;
    const bf16* bg = vb + ((size_t)k2 * HW_ + pix0) * 256 + c0;
#pragma unroll
    for (int i = 0; i < 2; ++i) {                       // stage A: 128 rows x 64B
      int t = tid + i * 256;
      int row = t >> 2, seg = t & 3;
      *(uint4*)&As[0][row * 40 + seg * 8] =
          *(const uint4*)(ag + (size_t)row * 2304 + seg * 8);
    }
    {                                                   // stage B: 64 rows x 64B
      int row = tid >> 2, seg = tid & 3;
      *(uint4*)&Bs[0][row * 40 + seg * 8] =
          *(const uint4*)(bg + (size_t)row * 256 + seg * 8);
    }
    if (kc + 1 < 72) {
      int kn = kc + 1, k2n = kn >> 3, c0n = (kn & 7) * 32;
      __builtin_prefetch(ab + (size_t)k2n * 256 + c0n, 0, 1);
      __builtin_prefetch(vb + ((size_t)k2n * HW_ + pix0) * 256 + c0n, 0, 1);
    }
    __syncthreads();
#endif
    // fragments per documented 16-bit A layout: lane holds M=lane%16,
    // K = (lane/16)*8 + {0..7} (VGPR0-3) and 16 + (lane/16)*8 + {0..7} (VGPR4-7)
    v16bf af[2], bfr[2];
#pragma unroll
    for (int i = 0; i < 2; ++i) {
      const bf16* ap = &As[cur][((m0 + i) * 16 + lm) * 40 + lh * 8];
      v8bf alo = *(const v8bf*)ap;
      v8bf ahi = *(const v8bf*)(ap + 16);
      af[i] = __builtin_shufflevector(alo, ahi, 0,1,2,3,4,5,6,7,8,9,10,11,12,13,14,15);
      const bf16* bp = &Bs[cur][((n0 + i) * 16 + lm) * 40 + lh * 8];
      v8bf blo = *(const v8bf*)bp;
      v8bf bhi = *(const v8bf*)(bp + 16);
      bfr[i] = __builtin_shufflevector(blo, bhi, 0,1,2,3,4,5,6,7,8,9,10,11,12,13,14,15);
    }
#pragma unroll
    for (int i = 0; i < 2; ++i)
#pragma unroll
      for (int j = 0; j < 2; ++j)
        acc[i][j] = __builtin_amdgcn_wmma_f32_16x16x32_bf16(
            false, af[i], false, bfr[j], (short)0, acc[i][j], false, false);
#if !USE_TDM
    __syncthreads();
#endif
  }

  // Epilogue: C/D layout — VGPR r, lane l holds (M = (l/16)*8 + r, N = l%16)
#pragma unroll
  for (int i = 0; i < 2; ++i) {
#pragma unroll
    for (int j = 0; j < 2; ++j) {
      int pix = pix0 + (n0 + j) * 16 + lm;
#pragma unroll
      for (int r = 0; r < 8; ++r) {
        int oc = oc0 + (m0 + i) * 16 + lh * 8 + r;
        out[((size_t)b * 256 + oc) * HW_ + pix] = acc[i][j][r] + bias[oc];
      }
    }
  }
}

// ---------------------------------------------------------------------------
extern "C" void kernel_launch(void* const* d_in, const int* in_sizes, int n_in,
                              void* d_out, int out_size, void* d_ws, size_t ws_size,
                              hipStream_t stream) {
  (void)in_sizes; (void)n_in; (void)out_size; (void)ws_size;
  const float* x    = (const float*)d_in[0];
  const float* em   = (const float*)d_in[1];
  const float* pw   = (const float*)d_in[2];
  const float* pb   = (const float*)d_in[3];
  const float* wgt  = (const float*)d_in[4];
  const float* bias = (const float*)d_in[5];
  float* out = (float*)d_out;

  char* ws = (char*)d_ws;
  size_t off = 0;
  auto alloc = [&](size_t bytes) {
    void* p = ws + off;
    off = (off + bytes + 255) & ~(size_t)255;
    return p;
  };
  float*  xt   = (float*) alloc(sizeof(float) * (size_t)B_ * HW_ * C_);   // 16.8 MB
  bf16*   abf  = (bf16*)  alloc(sizeof(bf16)  * 256ull * 2304);           // 1.2 MB
  int4*   idxw = (int4*)  alloc(sizeof(int4)  * (size_t)NPT_);            // 2.4 MB
  float4* wgtw = (float4*)alloc(sizeof(float4)* (size_t)NPT_);            // 2.4 MB
  bf16*   vws  = (bf16*)  alloc(sizeof(bf16)  * (size_t)NPT_ * 256);      // 75.5 MB

  xpose_kernel <<<2048,  256, 0, stream>>>(x, xt);
  wprep_kernel <<<2304,  256, 0, stream>>>(wgt, abf);
  head_kernel  <<<64,    256, 0, stream>>>(x, em, pw, pb, idxw, wgtw);
  vbuild_kernel<<<18432, 256, 0, stream>>>(xt, idxw, wgtw, vws);
  dcn_gemm_kernel<<<512, 256, 0, stream>>>(abf, vws, bias, out);
}
